// Attention_28071906246667
// MI455X (gfx1250) — matrix operand order repaired and checked by reading, not compiled, not run on hardware
//
#include <hip/hip_runtime.h>
#include <math.h>

// Problem dims
#define NB 16
#define CD 512
#define HW 3136
#define DD 512

typedef __attribute__((ext_vector_type(16))) __bf16       v16bf;
typedef __attribute__((ext_vector_type(8)))  float        v8f;
typedef __attribute__((ext_vector_type(4)))  unsigned int u32x4;
typedef __attribute__((ext_vector_type(8)))  int          i32x8;
typedef __attribute__((ext_vector_type(4)))  int          i32x4;

union FragAB {
    v16bf          v;
    unsigned short u[16];
    u32x4          q[2];
};

__device__ __forceinline__ unsigned short f2bf(float f) {
    unsigned int x = __float_as_uint(f);
    unsigned int r = (x + 0x7FFFu + ((x >> 16) & 1u)) >> 16;  // RNE
    return (unsigned short)r;
}

__device__ __forceinline__ v8f wmma_bf16(const FragAB& a, const FragAB& b, v8f c) {
    return __builtin_amdgcn_wmma_f32_16x16x32_bf16(
        /*neg_a=*/false, a.v, /*neg_b=*/false, b.v,
        /*c_mod=*/(short)0, c, /*reuse_a=*/false, /*reuse_b=*/false);
}

// TDM: DMA a [32 rows x 64 cols] bf16 tile (row stride = HW elements) from
// global into LDS at lds_addr. D# per CDNA5 ISA ch.8 (group0/group1 bitfields).
__device__ __forceinline__ void tdm_load_slab(unsigned int lds_addr,
                                              const unsigned short* gptr) {
    const unsigned long long ga = (unsigned long long)(const void*)gptr;
    u32x4 g0;
    g0[0] = 1u;                        // count=1, user mode
    g0[1] = lds_addr;                  // lds_addr (bytes)
    g0[2] = (unsigned int)ga;          // global_addr[31:0]
    g0[3] = (unsigned int)((ga >> 32) & 0x1FFFFFFull) | 0x80000000u;
                                       // global_addr[56:32] | type=2
    i32x8 g1;
    g1[0] = (int)(1u << 16);                              // data_size = 2 bytes
    g1[1] = (int)(((unsigned)HW & 0xFFFFu) << 16);        // tensor_dim0 lo16
    g1[2] = (int)(((unsigned)HW >> 16) |
                  (((unsigned)DD & 0xFFFFu) << 16));      // dim0 hi | dim1 lo
    g1[3] = (int)(((unsigned)DD >> 16) | (64u << 16));    // dim1 hi | tile0=64
    g1[4] = (int)32u;                                     // tile1=32, tile2=0
    g1[5] = (int)(unsigned)HW;                            // dim0_stride lo32
    g1[6] = 0;                                            // stride hi | d1s lo
    g1[7] = 0;
    i32x4 z4 = {0, 0, 0, 0};
    i32x8 z8 = {0, 0, 0, 0, 0, 0, 0, 0};
    __builtin_amdgcn_tensor_load_to_lds(g0, g1, z4, z4, z8, 0);
}

// ---------------------------------------------------------------------------
// Kernel 1: convert Wq/Wk/Wv f32 [D,C] -> bf16 (same layout, packed [3][D][C])
// ---------------------------------------------------------------------------
__global__ void wconvert_kernel(const float* __restrict__ Wq,
                                const float* __restrict__ Wk,
                                const float* __restrict__ Wv,
                                unsigned short* __restrict__ Wb) {
    const int total = DD * CD;
    int i = blockIdx.x * blockDim.x + threadIdx.x;
    if (i < total) {
        Wb[i]             = f2bf(Wq[i]);
        Wb[i + total]     = f2bf(Wk[i]);
        Wb[i + 2 * total] = f2bf(Wv[i]);
    }
}

// ---------------------------------------------------------------------------
// Kernel 2: transpose + convert: batch_flat f32 [n][c][s] -> XT bf16 [n][s][c]
// 64x64 tiles via padded LDS. grid = (HW/64=49, C/64=8, N=16), block 256.
// ---------------------------------------------------------------------------
__global__ void xpose_kernel(const float* __restrict__ bf,
                             unsigned short* __restrict__ XT) {
    __shared__ unsigned short tile[64][66];  // pad: 66 halfs = 33 dwords/row
    const int s0 = blockIdx.x * 64;
    const int c0 = blockIdx.y * 64;
    const int n  = blockIdx.z;
    const float* src = bf + (size_t)n * CD * HW;
    unsigned short* dst = XT + (size_t)n * HW * CD;

    const int tid = threadIdx.x;
    const int sl  = tid & 63;
    const int cl  = tid >> 6;  // 0..3
#pragma unroll
    for (int i = 0; i < 16; ++i) {
        const int c = cl + i * 4;
        tile[sl][c] = f2bf(src[(size_t)(c0 + c) * HW + s0 + sl]);
    }
    __syncthreads();
    const int cl2 = tid & 63;
    const int sl2 = tid >> 6;  // 0..3
#pragma unroll
    for (int i = 0; i < 16; ++i) {
        const int s = sl2 + i * 4;
        dst[(size_t)(s0 + s) * CD + c0 + cl2] = tile[s][cl2];
    }
}

// ---------------------------------------------------------------------------
// Kernel 3: QKV projection, stored TRANSPOSED (bf16):
//   YT[p][n][d][s] = sum_c W_p[d][c] * XT[n][s][c] + bias_p[d]
// A = W rows (contiguous over c), B[k=c][col=s] = XT[s][c] (contiguous over c)
// block = 8 waves: 128 d x 64 s ; wave = 16 d x 64 s (4 tiles)
// grid = (HW/64=49, D/128=4, N*3=48)
// ---------------------------------------------------------------------------
__global__ void qkv_kernel(const unsigned short* __restrict__ XT,
                           const unsigned short* __restrict__ Wb,
                           const float* __restrict__ bq,
                           const float* __restrict__ bk,
                           const float* __restrict__ bv,
                           unsigned short* __restrict__ QKVT) {
    const int p = blockIdx.z % 3;
    const int n = blockIdx.z / 3;
    const unsigned short* W = Wb + (size_t)p * DD * CD;
    const float* bias = (p == 0) ? bq : (p == 1) ? bk : bv;
    unsigned short* Y = QKVT + ((size_t)p * NB + n) * (size_t)DD * HW;
    const unsigned short* X = XT + (size_t)n * HW * CD;

    const int wv   = threadIdx.x >> 5;
    const int lane = threadIdx.x & 31;
    const int g    = lane >> 4;
    const int ln   = lane & 15;
    const int d0   = blockIdx.y * 128 + wv * 16;
    const int s0   = blockIdx.x * 64;

    v8f acc[4] = {{0,0,0,0,0,0,0,0}, {0,0,0,0,0,0,0,0},
                  {0,0,0,0,0,0,0,0}, {0,0,0,0,0,0,0,0}};

    for (int kk = 0; kk < CD; kk += 32) {
        FragAB a;
        const unsigned short* ar = W + (size_t)(d0 + ln) * CD + kk + g * 8;
        a.q[0] = *(const u32x4*)(ar);
        a.q[1] = *(const u32x4*)(ar + 16);

        FragAB b[4];
#pragma unroll
        for (int t = 0; t < 4; ++t) {
            const unsigned short* br =
                X + (size_t)(s0 + t * 16 + ln) * CD + kk + g * 8;
            b[t].q[0] = *(const u32x4*)(br);
            b[t].q[1] = *(const u32x4*)(br + 16);
        }
#pragma unroll
        for (int t = 0; t < 4; ++t)
            acc[t] = wmma_bf16(a, b[t], acc[t]);
    }

#pragma unroll
    for (int r = 0; r < 8; ++r) {
        const int d = d0 + r + 8 * g;
        const float bb = bias[d];
#pragma unroll
        for (int t = 0; t < 4; ++t)
            Y[(size_t)d * HW + s0 + t * 16 + ln] = f2bf(acc[t][r] + bb);
    }
}

// ---------------------------------------------------------------------------
// Kernel 4: scores  S[n][d][e] = sum_s QT[d][s] * KT[e][s]   (f32 out)
// grid = (D/64=8, D/128=4, N=16)
// ---------------------------------------------------------------------------
__global__ void scores_kernel(const unsigned short* __restrict__ QT,
                              const unsigned short* __restrict__ KT,
                              float* __restrict__ S) {
    const int n = blockIdx.z;
    const unsigned short* Q = QT + (size_t)n * DD * HW;
    const unsigned short* K = KT + (size_t)n * DD * HW;
    float* Sn = S + (size_t)n * DD * DD;

    const int wv   = threadIdx.x >> 5;
    const int lane = threadIdx.x & 31;
    const int g    = lane >> 4;
    const int ln   = lane & 15;
    const int d0   = blockIdx.y * 128 + wv * 16;
    const int e0   = blockIdx.x * 64;

    v8f acc[4] = {{0,0,0,0,0,0,0,0}, {0,0,0,0,0,0,0,0},
                  {0,0,0,0,0,0,0,0}, {0,0,0,0,0,0,0,0}};

    for (int s0 = 0; s0 < HW; s0 += 32) {
        FragAB a;
        const unsigned short* ar = Q + (size_t)(d0 + ln) * HW + s0;
        a.q[0] = *(const u32x4*)(ar + g * 8);
        a.q[1] = *(const u32x4*)(ar + 16 + g * 8);

        FragAB b[4];
#pragma unroll
        for (int t = 0; t < 4; ++t) {
            const unsigned short* br =
                K + (size_t)(e0 + t * 16 + ln) * HW + s0 + g * 8;
            b[t].q[0] = *(const u32x4*)(br);
            b[t].q[1] = *(const u32x4*)(br + 16);
        }
#pragma unroll
        for (int t = 0; t < 4; ++t)
            acc[t] = wmma_bf16(a, b[t], acc[t]);
    }

#pragma unroll
    for (int r = 0; r < 8; ++r) {
        const int d = d0 + r + 8 * g;
#pragma unroll
        for (int t = 0; t < 4; ++t)
            Sn[(size_t)d * DD + e0 + t * 16 + ln] = acc[t][r];
    }
}

// ---------------------------------------------------------------------------
// Kernel 5: row softmax over e (512) ; f32 in -> bf16 out. 1 block per row.
// ---------------------------------------------------------------------------
__global__ void softmax_kernel(const float* __restrict__ S,
                               unsigned short* __restrict__ ATT) {
    const int row = blockIdx.x;  // n*D + d
    const float* src = S + (size_t)row * DD;
    unsigned short* dst = ATT + (size_t)row * DD;

    const int t    = threadIdx.x;
    const int wv   = t >> 5;
    const int lane = t & 31;

    float v0 = src[t];
    float v1 = src[t + 256];
    float m = fmaxf(v0, v1);
#pragma unroll
    for (int off = 16; off >= 1; off >>= 1)
        m = fmaxf(m, __shfl_xor(m, off, 32));

    __shared__ float redm[8];
    __shared__ float reds[8];
    if (lane == 0) redm[wv] = m;
    __syncthreads();
    m = redm[0];
#pragma unroll
    for (int i = 1; i < 8; ++i) m = fmaxf(m, redm[i]);

    float e0 = __expf(v0 - m);
    float e1 = __expf(v1 - m);
    float s = e0 + e1;
#pragma unroll
    for (int off = 16; off >= 1; off >>= 1)
        s += __shfl_xor(s, off, 32);
    if (lane == 0) reds[wv] = s;
    __syncthreads();
    float tot = 0.f;
#pragma unroll
    for (int i = 0; i < 8; ++i) tot += reds[i];
    const float inv = 1.0f / tot;

    dst[t]       = f2bf(e0 * inv);
    dst[t + 256] = f2bf(e1 * inv);
}

// ---------------------------------------------------------------------------
// Kernel 6: out[n][d][s] = sum_e att[d][e] * VT[e][s]   (f32 out)
// A = att rows (contiguous over e).
// B slab [32e x 64s] DMA'd into LDS by the Tensor Data Mover, DOUBLE-BUFFERED:
// wave 0 issues the load for slab p^1 at the top of step i, then waits
// tensorcnt<=1 (TDM completes in-order per wave => the slab we consume is
// done) so the new DMA overlaps this step's fragment reads + 4 WMMAs.
// grid = (HW/64=49, D/128=4, N=16)
// ---------------------------------------------------------------------------
__global__ void out_kernel(const unsigned short* __restrict__ ATT,
                           const unsigned short* __restrict__ VT,
                           float* __restrict__ out) {
    __shared__ unsigned short slab[2][32 * 64];  // [buf][e_local][s_local]

    const int n = blockIdx.z;
    const unsigned short* A = ATT + (size_t)n * DD * DD;
    const unsigned short* V = VT + (size_t)n * DD * HW;
    float* O = out + (size_t)n * DD * HW;

    const int wv   = threadIdx.x >> 5;
    const int lane = threadIdx.x & 31;
    const int g    = lane >> 4;
    const int ln   = lane & 15;
    const int d0   = blockIdx.y * 128 + wv * 16;
    const int s0   = blockIdx.x * 64;

    const unsigned int lbase0 =
        (unsigned int)(unsigned long long)(const void*)&slab[0][0];
    const unsigned int lbase1 =
        (unsigned int)(unsigned long long)(const void*)&slab[1][0];

    v8f acc[4] = {{0,0,0,0,0,0,0,0}, {0,0,0,0,0,0,0,0},
                  {0,0,0,0,0,0,0,0}, {0,0,0,0,0,0,0,0}};

    if (wv == 0) tdm_load_slab(lbase0, V + s0);  // prologue: kk = 0 -> buf 0

    for (int kki = 0; kki < 16; ++kki) {
        const int kk = kki * 32;
        if (wv == 0) {
            if (kki < 15) {
                // prefetch next k-slab into the other buffer, then wait only
                // for the OLDER in-flight load (the one we consume this step)
                tdm_load_slab((kki & 1) ? lbase0 : lbase1,
                              V + (size_t)(kk + 32) * HW + s0);
                __builtin_amdgcn_s_wait_tensorcnt(1);
            } else {
                __builtin_amdgcn_s_wait_tensorcnt(0);
            }
        }
        __syncthreads();  // slab[kki&1] visible to all waves

        const unsigned short* sl = slab[kki & 1];

        FragAB a;
        const unsigned short* ar = A + (size_t)(d0 + ln) * DD + kk + g * 8;
        a.q[0] = *(const u32x4*)(ar);
        a.q[1] = *(const u32x4*)(ar + 16);

        FragAB b[4];
#pragma unroll
        for (int i = 0; i < 16; ++i) {
            const int row = ((i & 7) + g * 8 + (i >> 3) * 16) * 64;
#pragma unroll
            for (int t = 0; t < 4; ++t)
                b[t].u[i] = sl[row + t * 16 + ln];
        }
#pragma unroll
        for (int t = 0; t < 4; ++t)
            acc[t] = wmma_bf16(a, b[t], acc[t]);

        __syncthreads();  // all waves done reading before buffer is rewritten
    }

#pragma unroll
    for (int r = 0; r < 8; ++r) {
        const int d = d0 + r + 8 * g;
#pragma unroll
        for (int t = 0; t < 4; ++t)
            O[(size_t)d * HW + s0 + t * 16 + ln] = acc[t][r];
    }
}

// ---------------------------------------------------------------------------
extern "C" void kernel_launch(void* const* d_in, const int* in_sizes, int n_in,
                              void* d_out, int out_size, void* d_ws, size_t ws_size,
                              hipStream_t stream) {
    (void)in_sizes; (void)n_in; (void)out_size; (void)ws_size;

    const float* bf = (const float*)d_in[0];  // [N,C,HW]
    const float* Wq = (const float*)d_in[1];  // [D,C]
    const float* bq = (const float*)d_in[2];
    const float* Wk = (const float*)d_in[3];
    const float* bk = (const float*)d_in[4];
    const float* Wv = (const float*)d_in[5];
    const float* bv = (const float*)d_in[6];
    float* out = (float*)d_out;               // [N,D,HW]

    // Workspace layout (bf16 stored as ushort):
    //   Wb   : 3*D*C halfs
    //   XT   : N*HW*C halfs      (x transposed+converted)
    //   QKVT : 3*N*D*HW halfs    (Q^T, K^T, V^T)
    //   S    : N*D*D floats
    //   ATT  : N*D*D halfs
    const size_t DC  = (size_t)DD * CD;
    const size_t SC  = (size_t)HW * CD;
    const size_t DHW = (size_t)DD * HW;
    unsigned short* Wb   = (unsigned short*)d_ws;
    unsigned short* XT   = Wb + 3 * DC;
    unsigned short* QKVT = XT + (size_t)NB * SC;
    unsigned short* QT   = QKVT;
    unsigned short* KT   = QKVT + (size_t)NB * DHW;
    unsigned short* VT   = QKVT + 2 * (size_t)NB * DHW;
    float*          S    = (float*)(QKVT + 3 * (size_t)NB * DHW);
    unsigned short* ATT  = (unsigned short*)(S + (size_t)NB * DD * DD);

    wconvert_kernel<<<(DD * CD + 255) / 256, 256, 0, stream>>>(Wq, Wk, Wv, Wb);

    xpose_kernel<<<dim3(HW / 64, CD / 64, NB), 256, 0, stream>>>(bf, XT);

    qkv_kernel<<<dim3(HW / 64, DD / 128, NB * 3), 256, 0, stream>>>(
        XT, Wb, bq, bk, bv, QKVT);

    scores_kernel<<<dim3(DD / 64, DD / 128, NB), 256, 0, stream>>>(QT, KT, S);

    softmax_kernel<<<NB * DD, 256, 0, stream>>>(S, ATT);

    out_kernel<<<dim3(HW / 64, DD / 128, NB), 256, 0, stream>>>(ATT, VT, out);
}
